// KnowledgeLevel_22814866276942
// MI455X (gfx1250) — matrix operand order repaired and checked by reading, not compiled, not run on hardware
//
#include <hip/hip_runtime.h>
#include <math.h>

// ---- problem constants (fixed by the reference harness) ----
#define TT   20
#define LL   2000
#define NT   1000
#define KC   256
#define DD   5
#define MROWS (TT * LL)          // 40000
#define MTILES (MROWS / 16)      // 2500
#define NGROUPS 16               // 16 groups of up-to-4 n-tiles (64 cols each)
#define TASKS (MTILES * NGROUPS) // 40000 wave-tasks
#define WAVES_PER_BLOCK 8
#define GEMM_BLOCKS (TASKS / WAVES_PER_BLOCK) // 5000

// log-prob constant: -ln(0.1) - 0.5*ln(2*pi)
#define LP_CONST 1.3836465597893728f

typedef __attribute__((ext_vector_type(16))) __bf16 v16bf;
typedef __attribute__((ext_vector_type(8)))  __bf16 v8bf;
typedef __attribute__((ext_vector_type(8)))  float  v8f;

__device__ __forceinline__ unsigned short f2bf(float f) {
    unsigned int u = __float_as_uint(f);
    u += 0x7FFFu + ((u >> 16) & 1u);   // round-to-nearest-even
    return (unsigned short)(u >> 16);
}

// ---- kernel 1: column sums of neighbor_weights (sum over axis 0) ----
__global__ void colsum_kernel(const float* __restrict__ nw, float* __restrict__ colsum) {
    int j = blockIdx.x * blockDim.x + threadIdx.x;
    if (j >= NT) return;
    float s = 0.0f;
    for (int i = 0; i < NT; ++i) s += nw[(size_t)i * NT + j];
    colsum[j] = s;
}

// ---- kernel 2: V_j = V * colsum + 0.1*noise, packed bf16 (B matrix, K-contiguous) ----
__global__ void vj_kernel(const float* __restrict__ V, const float* __restrict__ noise,
                          const float* __restrict__ colsum, unsigned short* __restrict__ Vj) {
    int j = blockIdx.x;       // 0..NT-1
    int k = threadIdx.x;      // 0..KC-1
    size_t idx = (size_t)j * KC + k;
    float v = V[idx] * colsum[j] + 0.1f * noise[idx];
    Vj[idx] = f2bf(v);
}

// ---- kernel 3: U_emb[m,k] = dot(U[m,k,:], w) + b, packed bf16 (A matrix) ----
__global__ void uemb_kernel(const float* __restrict__ U, const float* __restrict__ w,
                            const float* __restrict__ b, unsigned short* __restrict__ Uemb) {
    __shared__ float sw[8];
    if (threadIdx.x < DD) sw[threadIdx.x] = w[threadIdx.x];
    if (threadIdx.x == DD) sw[DD] = b[0];
    __syncthreads();
    int m = blockIdx.x;       // 0..MROWS-1
    int k = threadIdx.x;      // 0..KC-1
    const float* p = U + ((size_t)m * KC + k) * DD;
    float acc = sw[DD];
#pragma unroll
    for (int d = 0; d < DD; ++d) acc += p[d] * sw[d];
    Uemb[(size_t)m * KC + k] = f2bf(acc);
}

// ---- kernel 4: zero the scalar output (harness poisons d_out) ----
__global__ void zero_kernel(float* out) {
    if (threadIdx.x == 0 && blockIdx.x == 0) out[0] = 0.0f;
}

// ---- per-wave 16(M) x NTILES*16(N) tile: double-buffered bf16 WMMA + fused epilogue ----
template <int NTILES>
__device__ __forceinline__ float tile_compute(const __bf16* __restrict__ Uemb,
                                              const __bf16* __restrict__ Vj,
                                              const float* __restrict__ R,
                                              const int*   __restrict__ C,
                                              int m0, int nBase, int lane) {
    // A fragment (16-bit A 16x32): lanes 0-15 row=lane, K=0..15;
    //                              lanes 16-31 row=lane-16, K=16..31
    const int rowA  = m0 + (lane & 15);
    const int koffA = (lane & 16) ? 16 : 0;
    // B fragment (16-bit B 32x16): lanes 0-15 col=lane, K=0..15; lanes 16-31 K=16..31
    const int nLocal = lane & 15;
    const int kbaseB = (lane & 16) ? 16 : 0;

    const __bf16* pa = Uemb + (size_t)rowA * KC + koffA;
    const __bf16* pb[NTILES];
#pragma unroll
    for (int nt = 0; nt < NTILES; ++nt) {
        int col = nBase + nt * 16 + nLocal;
        if (col >= NT) col = NT - 1;           // clamp loads; masked in epilogue
        pb[nt] = Vj + (size_t)col * KC + kbaseB;
    }

    union AFrag { v16bf v; v8bf h[2]; };
    AFrag A[2];
    v16bf Bf[2][NTILES];

    // prologue: kt = 0
    A[0].h[0] = *(const v8bf*)(pa + 0);
    A[0].h[1] = *(const v8bf*)(pa + 8);
#pragma unroll
    for (int nt = 0; nt < NTILES; ++nt) Bf[0][nt] = *(const v16bf*)(pb[nt]);

    v8f acc[NTILES];
#pragma unroll
    for (int i = 0; i < NTILES; ++i)
#pragma unroll
        for (int e = 0; e < 8; ++e) acc[i][e] = 0.0f;

#pragma unroll
    for (int kt = 0; kt < 8; ++kt) {
        const int cur = kt & 1, nxt = cur ^ 1;
        if (kt < 7) {   // prefetch next k-step while this step's WMMAs run
            const int k0n = (kt + 1) * 32;
            A[nxt].h[0] = *(const v8bf*)(pa + k0n);
            A[nxt].h[1] = *(const v8bf*)(pa + k0n + 8);
#pragma unroll
            for (int nt = 0; nt < NTILES; ++nt)
                Bf[nxt][nt] = *(const v16bf*)(pb[nt] + k0n);
        }
#pragma unroll
        for (int nt = 0; nt < NTILES; ++nt)
            acc[nt] = __builtin_amdgcn_wmma_f32_16x16x32_bf16(
                /*neg_a=*/false, A[cur].v, /*neg_b=*/false, Bf[cur][nt],
                /*c_mod=*/(short)0, acc[nt], /*reuse_a=*/false, /*reuse_b=*/false);
    }

    // Epilogue: C/D f32 16x16 layout -> VGPR r holds (M=r | r+8, N=lane&15)
    float lsum = 0.0f;
    const int hi = (lane & 16) ? 8 : 0;
#pragma unroll
    for (int nt = 0; nt < NTILES; ++nt) {
        const int col = nBase + nt * 16 + nLocal;
        if (col < NT) {
#pragma unroll
            for (int r = 0; r < 8; ++r) {
                const int row = m0 + r + hi;
                const float x  = acc[nt][r];
                const float mu = 1.0f / (1.0f + __expf(-x));
                const size_t idx = (size_t)row * NT + col;
                const float Rv = R[idx];
                const int   Cv = C[idx];
                if (Cv == 1) {
                    const float z = (Rv - mu) * 10.0f;   // / SCALE_R
                    lsum += -0.5f * z * z + LP_CONST;
                }
            }
        }
    }
    return lsum;
}

// ---- kernel 5: fused bf16 WMMA GEMM + sigmoid + masked Gaussian log-prob + reduce ----
__global__ void __launch_bounds__(256)
fused_gemm_kernel(const unsigned short* __restrict__ UembU,
                  const unsigned short* __restrict__ VjU,
                  const float* __restrict__ R,
                  const int*   __restrict__ C,
                  float* __restrict__ out) {
    const __bf16* Uemb = (const __bf16*)UembU;
    const __bf16* Vj   = (const __bf16*)VjU;

    const int lane = threadIdx.x & 31;
    // readfirstlane -> wave index is an SGPR, so tile coords are scalar and the
    // partial-group branch below is a scalar branch (EXEC stays all-ones for WMMA).
    const int wave = __builtin_amdgcn_readfirstlane((int)(threadIdx.x >> 5));
    const int task = blockIdx.x * WAVES_PER_BLOCK + wave;   // < TASKS exactly
    const int m0    = (task >> 4) * 16;     // 2500 m-tiles
    const int nBase = (task & 15) * 64;     // 16 n-groups of 64 cols

    float lsum;
    if (nBase + 64 <= NT)   // groups 0..14: 4 full n-tiles
        lsum = tile_compute<4>(Uemb, Vj, R, C, m0, nBase, lane);
    else                    // group 15: tiles at 960/976/992 (992 partial, clamped+masked)
        lsum = tile_compute<3>(Uemb, Vj, R, C, m0, nBase, lane);

    // Block reduction + single atomic per block
    __shared__ float red[256];
    red[threadIdx.x] = lsum;
    __syncthreads();
#pragma unroll
    for (int s = 128; s > 0; s >>= 1) {
        if (threadIdx.x < s) red[threadIdx.x] += red[threadIdx.x + s];
        __syncthreads();
    }
    if (threadIdx.x == 0) atomicAdd(out, red[0]);
}

extern "C" void kernel_launch(void* const* d_in, const int* in_sizes, int n_in,
                              void* d_out, int out_size, void* d_ws, size_t ws_size,
                              hipStream_t stream) {
    // setup_inputs() order: Q, V, R, neighbor_weights, U, U_emb_w, U_emb_b, noise, C
    const float* V     = (const float*)d_in[1];
    const float* R     = (const float*)d_in[2];
    const float* nw    = (const float*)d_in[3];
    const float* U     = (const float*)d_in[4];
    const float* w     = (const float*)d_in[5];
    const float* b     = (const float*)d_in[6];
    const float* noise = (const float*)d_in[7];
    const int*   C     = (const int*)d_in[8];
    float* out = (float*)d_out;

    char* ws = (char*)d_ws;
    float*          colsum = (float*)ws;                               // 4 KB
    unsigned short* Vj     = (unsigned short*)(ws + 4096);             // 512 KB
    unsigned short* Uemb   = (unsigned short*)(ws + 4096 + 524288);    // 20.48 MB

    zero_kernel<<<1, 1, 0, stream>>>(out);
    colsum_kernel<<<(NT + 255) / 256, 256, 0, stream>>>(nw, colsum);
    vj_kernel<<<NT, KC, 0, stream>>>(V, noise, colsum, Vj);
    uemb_kernel<<<MROWS, KC, 0, stream>>>(U, w, b, Uemb);
    fused_gemm_kernel<<<GEMM_BLOCKS, 256, 0, stream>>>(Uemb, Vj, R, C, out);
}
